// Seq2Seq_15771119911401
// MI455X (gfx1250) — matrix operand order repaired and checked by reading, not compiled, not run on hardware
//
#include <hip/hip_runtime.h>
#include <hip/hip_bf16.h>
#include <stdint.h>
#include <stddef.h>

// ---------------- problem constants ----------------
#define BB    64      // batch
#define SS    64      // source length (encoder steps)
#define TT    64      // target length (decoder steps = TT-1)
#define HH    1024    // hidden == embed dim
#define H3    3072    // 3*H (GRU gates)
#define VOUTD 32000   // output vocab
#define NLAY  2
#define WAVES 4       // waves per workgroup in WMMA kernels (128 threads)
#define NT    4       // N tiles per wave in the generic GEMM (A-fragment reuse)

typedef __attribute__((ext_vector_type(16))) __bf16 v16bf;
typedef __attribute__((ext_vector_type(8)))  float  v8f;

// ---- fp32 -> bf16 round-to-nearest-even ----
__device__ __forceinline__ unsigned short f2bf(float x) {
  union { float f; unsigned int u; } c; c.f = x;
  unsigned int u = c.u;
  u += 0x7FFFu + ((u >> 16) & 1u);
  return (unsigned short)(u >> 16);
}

// ---- fast gate nonlinearities: v_exp_f32 + v_rcp_f32, no IEEE divide ----
__device__ __forceinline__ float fast_sigmoid(float x) {
  return __builtin_amdgcn_rcpf(1.0f + __expf(-x));
}
__device__ __forceinline__ float fast_tanh(float x) {
  // tanh(x) = 1 - 2/(e^{2x}+1)
  return 1.0f - 2.0f * __builtin_amdgcn_rcpf(__expf(2.0f * x) + 1.0f);
}

union FragU { unsigned int u[8]; v16bf v; };

// A-matrix (16x32 bf16) fragment, contiguous k order (same as load_b_frag):
// lane holds row m = lane&15; K = half*16 + 2v..2v+1 -> dwords [half*8, half*8+7].
// Collapses to two b128 loads (offset half*32 within the 64B k-slice).
__device__ __forceinline__ v16bf load_a_frag(const unsigned short* rowk, int half) {
  const unsigned int* p = (const unsigned int*)rowk;
  FragU fr;
#pragma unroll
  for (int v = 0; v < 8; ++v) {
    int pr = v + (half << 3);
    fr.u[v] = p[pr];
  }
  return fr.v;
}

// B-matrix (32x16 bf16) fragment: lane holds column n = lane&15 (== row n of W
// since B = W^T), K = half*16 + 2v..2v+1 -> dwords [half*8, half*8+7] (two b128s).
__device__ __forceinline__ v16bf load_b_frag(const unsigned short* rowk, int half) {
  const unsigned int* p = (const unsigned int*)rowk;
  FragU fr;
#pragma unroll
  for (int v = 0; v < 8; ++v) {
    fr.u[v] = p[v + (half << 3)];
  }
  return fr.v;
}

// ---------------- generic WMMA GEMM: C = A * W^T (+ epilogue) ----------------
// A: [M,K] bf16 row-major (lda), W: [N,K] bf16 row-major (ldw), C: f32.
// Each wave owns NT adjacent 16-wide N tiles: 1 A fragment feeds NT WMMAs/k-step.
// MODE 0: plain store C[m*ldc+n].
// MODE 1: logits scatter with bias: m = (t-1)*64 + b -> out[b*TT*VOUT + t*VOUT + n].
template <int MODE>
__launch_bounds__(32 * WAVES)
__global__ void wmma_gemm_kernel(const unsigned short* __restrict__ A, int lda,
                                 const unsigned short* __restrict__ W, int ldw,
                                 float* __restrict__ C, int ldc,
                                 const float* __restrict__ bias,
                                 int M, int N, int K) {
  const int lane = threadIdx.x & 31;
  const int wave = threadIdx.x >> 5;
  const int tn0  = (blockIdx.x * WAVES + wave) * NT;  // first N tile of this wave
  const int tm   = blockIdx.y;                        // M tile index
  if (tn0 * 16 >= N || tm * 16 >= M) return;          // wave-uniform -> EXEC stays full
  const int half = lane >> 4;
  const int l15  = lane & 15;

  const unsigned short* arow = A + (size_t)(tm * 16 + l15) * lda;
  const unsigned short* wrow[NT];
#pragma unroll
  for (int nt = 0; nt < NT; ++nt)
    wrow[nt] = W + (size_t)((tn0 + nt) * 16 + l15) * ldw;

  v8f acc[NT];
#pragma unroll
  for (int nt = 0; nt < NT; ++nt) acc[nt] = (v8f){};

  for (int k0 = 0; k0 < K; k0 += 32) {
    __builtin_prefetch(wrow[0] + k0 + 256, 0, 0);   // global_prefetch_b8 on weight stream
    v16bf a = load_a_frag(arow + k0, half);
#pragma unroll
    for (int nt = 0; nt < NT; ++nt) {
      v16bf b = load_b_frag(wrow[nt] + k0, half);
      acc[nt] = __builtin_amdgcn_wmma_f32_16x16x32_bf16(false, a, false, b,
                                                        (short)0, acc[nt], false, false);
    }
  }

#pragma unroll
  for (int nt = 0; nt < NT; ++nt) {
    const int n = (tn0 + nt) * 16 + l15;
#pragma unroll
    for (int v = 0; v < 8; ++v) {
      int m = tm * 16 + v + half * 8;
      float val = acc[nt][v];
      if (MODE == 0) {
        C[(size_t)m * ldc + n] = val;
      } else {
        int t = (m >> 6) + 1;          // decoder output slot (outputs[:,0] is zero)
        int b = m & 63;
        C[((size_t)b * TT + t) * (size_t)ldc + n] = val + bias[n];
      }
    }
  }
}

// ---------------- fused GRU step: gh-GEMM + gates + mask, one layer ----------------
// Wave computes the r/z/n 16x16 tiles for one (row-tile, hidden-tile) pair so the
// full gate math stays in registers. h is double-buffered (cur -> nxt) to avoid
// cross-workgroup read/write races on the recurrent input.
__launch_bounds__(32 * WAVES)
__global__ void gru_step_fused_kernel(
    const unsigned short* __restrict__ Hb_cur,  // [B,H] bf16   (GEMM A input)
    const float*          __restrict__ Hf_cur,  // [B,H] f32    (old hidden)
    const unsigned short* __restrict__ Whh,     // [3H,H] bf16  (layer base)
    const float*          __restrict__ gi,      // [B,3H] f32   (x@Wih^T, no bias)
    const float*          __restrict__ bih,     // [3H]
    const float*          __restrict__ bhh,     // [3H]
    float*                __restrict__ Hf_nxt,  // [B,H] f32 out
    unsigned short*       __restrict__ Hb_nxt,  // [B,H] bf16 out
    unsigned short*       __restrict__ Xnext,   // [B,H] bf16 out (unmasked h_new) or null
    const int*            __restrict__ lengths, // [B] encoder mask or null
    int t) {
  const int lane = threadIdx.x & 31;
  const int wave = threadIdx.x >> 5;
  const int jt   = blockIdx.x * WAVES + wave;   // hidden tile 0..H/16-1
  const int tm   = blockIdx.y;                  // batch tile 0..B/16-1
  const int half = lane >> 4;
  const int l15  = lane & 15;

  const unsigned short* arow = Hb_cur + (size_t)(tm * 16 + l15) * HH;
  const unsigned short* wr = Whh + (size_t)(           jt * 16 + l15) * HH;
  const unsigned short* wz = Whh + (size_t)(HH     +   jt * 16 + l15) * HH;
  const unsigned short* wn = Whh + (size_t)(2 * HH +   jt * 16 + l15) * HH;

  v8f ar = {}, az = {}, an = {};
  for (int k0 = 0; k0 < HH; k0 += 32) {
    // issue all four fragment loads, then the three dependent WMMAs
    v16bf a  = load_a_frag(arow + k0, half);
    v16bf br = load_b_frag(wr + k0, half);
    v16bf bz = load_b_frag(wz + k0, half);
    v16bf bn = load_b_frag(wn + k0, half);
    ar = __builtin_amdgcn_wmma_f32_16x16x32_bf16(false, a, false, br, (short)0, ar, false, false);
    az = __builtin_amdgcn_wmma_f32_16x16x32_bf16(false, a, false, bz, (short)0, az, false, false);
    an = __builtin_amdgcn_wmma_f32_16x16x32_bf16(false, a, false, bn, (short)0, an, false, false);
  }

  const int   j   = jt * 16 + l15;
  const float cr_ = bih[j], cz_ = bih[HH + j], cn_ = bih[2 * HH + j];
  const float br_ = bhh[j], bz_ = bhh[HH + j], bn_ = bhh[2 * HH + j];

#pragma unroll
  for (int v = 0; v < 8; ++v) {
    int m = tm * 16 + v + half * 8;               // batch row
    const float* gir = gi + (size_t)m * H3;
    float r  = fast_sigmoid(gir[j]      + cr_ + ar[v] + br_);
    float z  = fast_sigmoid(gir[HH + j] + cz_ + az[v] + bz_);
    float nn = fast_tanh(gir[2 * HH + j] + cn_ + r * (an[v] + bn_));
    float hold = Hf_cur[(size_t)m * HH + j];
    float hnew = (1.0f - z) * nn + z * hold;
    if (Xnext) Xnext[(size_t)m * HH + j] = f2bf(hnew);   // next layer sees unmasked h_new
    float hs = hnew;
    if (lengths && !(t < lengths[m])) hs = hold;          // packed-seq freeze
    Hf_nxt[(size_t)m * HH + j] = hs;
    Hb_nxt[(size_t)m * HH + j] = f2bf(hs);
  }
}

// ---------------- utility kernels ----------------
__global__ void f32_to_bf16_kernel(const float* __restrict__ src,
                                   unsigned short* __restrict__ dst, long long n) {
  long long i = (long long)blockIdx.x * blockDim.x + threadIdx.x;
  if (i < n) dst[i] = f2bf(src[i]);
}

// X[(t*B+b)*E + e] = table[tokens[b*tokStride + t]][e]  (row 0 of table is zero already)
__global__ void embed_gather_kernel(const int* __restrict__ tokens, int tokStride,
                                    const float* __restrict__ table,
                                    unsigned short* __restrict__ X,
                                    long long total) {
  long long i = (long long)blockIdx.x * blockDim.x + threadIdx.x;
  if (i >= total) return;
  int e = (int)(i % HH);
  long long row = i / HH;           // t*B + b
  int b = (int)(row % BB);
  int t = (int)(row / BB);
  int tok = tokens[b * tokStride + t];
  X[i] = f2bf(table[(size_t)tok * HH + e]);
}

__global__ void zero_f32_kernel(float* p, long long n) {
  long long i = (long long)blockIdx.x * blockDim.x + threadIdx.x;
  if (i < n) p[i] = 0.0f;
}
__global__ void zero_bf16_kernel(unsigned short* p, long long n) {
  long long i = (long long)blockIdx.x * blockDim.x + threadIdx.x;
  if (i < n) p[i] = 0;
}
// out[b, 0, :] = 0
__global__ void zero_out_t0_kernel(float* out) {
  long long i = (long long)blockIdx.x * blockDim.x + threadIdx.x;
  if (i >= (long long)BB * VOUTD) return;
  long long b = i / VOUTD, v = i % VOUTD;
  out[b * (long long)TT * VOUTD + v] = 0.0f;
}

// ---------------- host orchestration ----------------
extern "C" void kernel_launch(void* const* d_in, const int* in_sizes, int n_in,
                              void* d_out, int out_size, void* d_ws, size_t ws_size,
                              hipStream_t stream) {
  (void)in_sizes; (void)n_in; (void)out_size; (void)ws_size;
  const int*   src_tokens = (const int*)  d_in[0];
  const int*   src_len    = (const int*)  d_in[1];
  const int*   trg        = (const int*)  d_in[2];
  const float* emb_enc    = (const float*)d_in[3];
  const float* Wih_enc    = (const float*)d_in[4];
  const float* Whh_enc    = (const float*)d_in[5];
  const float* bih_enc    = (const float*)d_in[6];
  const float* bhh_enc    = (const float*)d_in[7];
  const float* emb_dec    = (const float*)d_in[8];
  const float* Wih_dec    = (const float*)d_in[9];
  const float* Whh_dec    = (const float*)d_in[10];
  const float* bih_dec    = (const float*)d_in[11];
  const float* bhh_dec    = (const float*)d_in[12];
  const float* Wout       = (const float*)d_in[13];
  const float* bout       = (const float*)d_in[14];
  float* out = (float*)d_out;

  // ---- workspace carve-up (~240 MB) ----
  char* ws = (char*)d_ws; size_t off = 0;
  auto alloc = [&](size_t bytes) -> void* {
    void* p = ws + off; off += (bytes + 255) & ~(size_t)255; return p;
  };
  unsigned short* wih_enc_b = (unsigned short*)alloc((size_t)NLAY * H3 * HH * 2);
  unsigned short* whh_enc_b = (unsigned short*)alloc((size_t)NLAY * H3 * HH * 2);
  unsigned short* wih_dec_b = (unsigned short*)alloc((size_t)NLAY * H3 * HH * 2);
  unsigned short* whh_dec_b = (unsigned short*)alloc((size_t)NLAY * H3 * HH * 2);
  unsigned short* wout_b    = (unsigned short*)alloc((size_t)VOUTD * HH * 2);
  unsigned short* xenc      = (unsigned short*)alloc((size_t)SS * BB * HH * 2);
  unsigned short* xdec      = (unsigned short*)alloc((size_t)(TT - 1) * BB * HH * 2);
  float*          gi_enc0   = (float*)alloc((size_t)SS * BB * H3 * 4);
  float*          gi_dec0   = (float*)alloc((size_t)(TT - 1) * BB * H3 * 4);
  float*          gi1       = (float*)alloc((size_t)BB * H3 * 4);
  float*          hf_all    = (float*)alloc((size_t)NLAY * 2 * BB * HH * 4);          // [layer][buf]
  unsigned short* hb_all    = (unsigned short*)alloc((size_t)NLAY * 2 * BB * HH * 2); // [layer][buf]
  unsigned short* x1        = (unsigned short*)alloc((size_t)BB * HH * 2);
  unsigned short* tops      = (unsigned short*)alloc((size_t)(TT - 1) * BB * HH * 2);

  auto hf = [&](int l, int buf) { return hf_all + ((size_t)l * 2 + buf) * BB * HH; };
  auto hb = [&](int l, int buf) { return hb_all + ((size_t)l * 2 + buf) * BB * HH; };

  const int TPB = 256;
  auto cdiv = [](long long a, long long b) { return (int)((a + b - 1) / b); };

  // ---- prep: weight conversion fp32 -> bf16 (weights then live in L2) ----
  f32_to_bf16_kernel<<<cdiv((long long)NLAY * H3 * HH, TPB), TPB, 0, stream>>>(Wih_enc, wih_enc_b, (long long)NLAY * H3 * HH);
  f32_to_bf16_kernel<<<cdiv((long long)NLAY * H3 * HH, TPB), TPB, 0, stream>>>(Whh_enc, whh_enc_b, (long long)NLAY * H3 * HH);
  f32_to_bf16_kernel<<<cdiv((long long)NLAY * H3 * HH, TPB), TPB, 0, stream>>>(Wih_dec, wih_dec_b, (long long)NLAY * H3 * HH);
  f32_to_bf16_kernel<<<cdiv((long long)NLAY * H3 * HH, TPB), TPB, 0, stream>>>(Whh_dec, whh_dec_b, (long long)NLAY * H3 * HH);
  f32_to_bf16_kernel<<<cdiv((long long)VOUTD * HH, TPB), TPB, 0, stream>>>(Wout, wout_b, (long long)VOUTD * HH);

  // ---- prep: embedding gathers (time-major rows t*B+b) ----
  embed_gather_kernel<<<cdiv((long long)SS * BB * HH, TPB), TPB, 0, stream>>>(
      src_tokens, SS, emb_enc, xenc, (long long)SS * BB * HH);
  embed_gather_kernel<<<cdiv((long long)(TT - 1) * BB * HH, TPB), TPB, 0, stream>>>(
      trg, TT, emb_dec, xdec, (long long)(TT - 1) * BB * HH);

  // ---- zero hidden state buffers and out[:,0,:] ----
  zero_f32_kernel<<<cdiv((long long)NLAY * 2 * BB * HH, TPB), TPB, 0, stream>>>(hf_all, (long long)NLAY * 2 * BB * HH);
  zero_bf16_kernel<<<cdiv((long long)NLAY * 2 * BB * HH, TPB), TPB, 0, stream>>>(hb_all, (long long)NLAY * 2 * BB * HH);
  zero_out_t0_kernel<<<cdiv((long long)BB * VOUTD, TPB), TPB, 0, stream>>>(out);

  // ---- batched layer-0 input-gate GEMMs (no recurrence dependence) ----
  wmma_gemm_kernel<0><<<dim3((H3 / 16) / (WAVES * NT), (SS * BB) / 16), 32 * WAVES, 0, stream>>>(
      xenc, HH, wih_enc_b, HH, gi_enc0, H3, nullptr, SS * BB, H3, HH);
  wmma_gemm_kernel<0><<<dim3((H3 / 16) / (WAVES * NT), ((TT - 1) * BB) / 16), 32 * WAVES, 0, stream>>>(
      xdec, HH, wih_dec_b, HH, gi_dec0, H3, nullptr, (TT - 1) * BB, H3, HH);

  const dim3 stepGrid((HH / 16) / WAVES, BB / 16);        // 16 x 4 blocks
  const dim3 gi1Grid((H3 / 16) / (WAVES * NT), BB / 16);  // 12 x 4 blocks
  int cur0 = 0, cur1 = 0;

  // ---- encoder recurrence (masked by src_lengths) ----
  for (int t = 0; t < SS; ++t) {
    gru_step_fused_kernel<<<stepGrid, 32 * WAVES, 0, stream>>>(
        hb(0, cur0), hf(0, cur0), whh_enc_b, gi_enc0 + (size_t)t * BB * H3,
        bih_enc, bhh_enc, hf(0, cur0 ^ 1), hb(0, cur0 ^ 1), x1, src_len, t);
    cur0 ^= 1;
    wmma_gemm_kernel<0><<<gi1Grid, 32 * WAVES, 0, stream>>>(
        x1, HH, wih_enc_b + (size_t)H3 * HH, HH, gi1, H3, nullptr, BB, H3, HH);
    gru_step_fused_kernel<<<stepGrid, 32 * WAVES, 0, stream>>>(
        hb(1, cur1), hf(1, cur1), whh_enc_b + (size_t)H3 * HH, gi1,
        bih_enc + H3, bhh_enc + H3, hf(1, cur1 ^ 1), hb(1, cur1 ^ 1),
        (unsigned short*)nullptr, src_len, t);
    cur1 ^= 1;
  }

  // ---- decoder recurrence (teacher forced; tops collected in bf16) ----
  for (int s = 0; s < TT - 1; ++s) {
    gru_step_fused_kernel<<<stepGrid, 32 * WAVES, 0, stream>>>(
        hb(0, cur0), hf(0, cur0), whh_dec_b, gi_dec0 + (size_t)s * BB * H3,
        bih_dec, bhh_dec, hf(0, cur0 ^ 1), hb(0, cur0 ^ 1), x1,
        (const int*)nullptr, 0);
    cur0 ^= 1;
    wmma_gemm_kernel<0><<<gi1Grid, 32 * WAVES, 0, stream>>>(
        x1, HH, wih_dec_b + (size_t)H3 * HH, HH, gi1, H3, nullptr, BB, H3, HH);
    gru_step_fused_kernel<<<stepGrid, 32 * WAVES, 0, stream>>>(
        hb(1, cur1), hf(1, cur1), whh_dec_b + (size_t)H3 * HH, gi1,
        bih_dec + H3, bhh_dec + H3, hf(1, cur1 ^ 1), hb(1, cur1 ^ 1),
        tops + (size_t)s * BB * HH, (const int*)nullptr, 0);
    cur1 ^= 1;
  }

  // ---- single batched output projection with bias + scatter to [B,T,V] ----
  wmma_gemm_kernel<1><<<dim3((VOUTD / 16) / (WAVES * NT), ((TT - 1) * BB) / 16), 32 * WAVES, 0, stream>>>(
      tops, HH, wout_b, HH, out, VOUTD, bout, (TT - 1) * BB, VOUTD, HH);
}
